// BQQLinear_82635170775232
// MI455X (gfx1250) — matrix operand order repaired and checked by reading, not compiled, not run on hardware
//
#include <hip/hip_runtime.h>
#include <hip/hip_bf16.h>

typedef __attribute__((ext_vector_type(8)))  float   v8f;
typedef __attribute__((ext_vector_type(8)))  __bf16  v8bf;
typedef __attribute__((ext_vector_type(16))) __bf16  v16bf;
typedef __attribute__((ext_vector_type(4)))  int     v4i;

union V16 { v16bf v; v8bf h[2]; };

__device__ inline unsigned short f32_to_bf16(float f) {
    union { float f; unsigned u; } x; x.f = f;
    unsigned r = x.u + 0x7fffu + ((x.u >> 16) & 1u);   // round-to-nearest-even
    return (unsigned short)(r >> 16);
}

// problem constants (from reference): p=2, j=64, k=64, m=64, l=8, n=64, B=512
constexpr int P = 2, J = 64, K = 64, M = 64, L = 8, N = 64;
constexpr int KN = K * N;     // 4096 = GEMM contraction dim
constexpr int JM = J * M;     // 4096 = GEMM output columns
constexpr int BATCH = 512;    // GEMM rows

// ---------------- async-to-LDS availability (device pass only) -------------
#if defined(__HIP_DEVICE_COMPILE__) && __has_builtin(__builtin_amdgcn_global_load_async_to_lds_b128)
#define HAVE_ASYNC 1
#else
#define HAVE_ASYNC 0
#endif

#if HAVE_ASYNC
#if __has_builtin(__builtin_amdgcn_s_wait_asynccnt)
#define WAIT_ASYNC(n) __builtin_amdgcn_s_wait_asynccnt((short)(n))
#else
#define WAIT_ASYNC(n) asm volatile("s_wait_asynccnt %0" ::"i"(n) : "memory")
#endif
#else
#define WAIT_ASYNC(n) do {} while (0)
#endif

// 128B (8 x b128) per-thread chunk copy: global -> LDS
__device__ __forceinline__ void stage_chunk(const unsigned short* gsrc,
                                            unsigned short* ldst) {
#if HAVE_ASYNC
    #pragma unroll
    for (int i = 0; i < 8; ++i) {
        __builtin_amdgcn_global_load_async_to_lds_b128(
            (__attribute__((address_space(1))) v4i*)(const void*)(gsrc + i * 8),
            (__attribute__((address_space(3))) v4i*)(void*)(ldst + i * 8),
            0, 0);
    }
#else
    const uint4* s = (const uint4*)gsrc;
    uint4* d = (uint4*)ldst;
    #pragma unroll
    for (int i = 0; i < 8; ++i) d[i] = s[i];
#endif
}

// ---------------------------------------------------------------------------
// Kernel 1: fold scales/signs/A/broadcast terms into one bf16 weight matrix
//   WT[col = j*64+m][Kidx = k*64+n]   (K contiguous -> contiguous B-fragments)
// ---------------------------------------------------------------------------
__global__ __launch_bounds__(256)
void build_w(const float* __restrict__ Y_fp,
             const float* __restrict__ Z_fp,
             const float* __restrict__ A,
             unsigned short* __restrict__ WT)
{
    const int k   = blockIdx.x;
    const int j   = blockIdx.y;
    const int tid = threadIdx.x;

    __shared__ float sY[P][M * L];   // sign(Y)  (64x8 per p)
    __shared__ float sZ[P][L * N];   // sign(Z)  (8x64 per p)
    __shared__ float sAbsY[P], sAbsZ[P];
    __shared__ float sA[P][4];

    if (tid < P) { sAbsY[tid] = 0.f; sAbsZ[tid] = 0.f; }
    if (tid < P * 4) {
        int p = tid >> 2, c = tid & 3;
        sA[p][c] = A[((size_t)(p * J + j) * K + k) * 4 + c];
    }
    __syncthreads();

    for (int p = 0; p < P; ++p) {
        const size_t base = ((size_t)(p * J + j) * K + k) * 512;
        float ay = 0.f, az = 0.f;
        #pragma unroll
        for (int e = 0; e < 2; ++e) {
            int i = tid * 2 + e;
            float y = Y_fp[base + i];
            float z = Z_fp[base + i];
            ay += fabsf(y); az += fabsf(z);
            sY[p][i] = (y > 0.f) ? 1.f : ((y < 0.f) ? -1.f : 0.f);
            sZ[p][i] = (z > 0.f) ? 1.f : ((z < 0.f) ? -1.f : 0.f);
        }
        atomicAdd(&sAbsY[p], ay);   // ds_add_f32 on LDS
        atomicAdd(&sAbsZ[p], az);
    }
    __syncthreads();

    const int m  = tid >> 2;
    const int nb = (tid & 3) * 16;

    float Ysc[P], Zsc[P], ysum[P];
    #pragma unroll
    for (int p = 0; p < P; ++p) {
        Ysc[p] = sAbsY[p] * (1.f / 512.f);
        Zsc[p] = sAbsZ[p] * (1.f / 512.f);
        float s = 0.f;
        #pragma unroll
        for (int l = 0; l < L; ++l) s += sY[p][m * L + l];
        ysum[p] = s;
    }
    float Bc = 0.f, Dc = 0.f;
    #pragma unroll
    for (int p = 0; p < P; ++p) { Bc += sA[p][1] * Ysc[p] * ysum[p]; Dc += sA[p][3]; }

    for (int nn = 0; nn < 16; ++nn) {
        const int n = nb + nn;
        float val = Bc + Dc;
        #pragma unroll
        for (int p = 0; p < P; ++p) {
            float inner = 0.f, zsum = 0.f;
            #pragma unroll
            for (int l = 0; l < L; ++l) {
                float zs = sZ[p][l * N + n];
                inner += sY[p][m * L + l] * zs;
                zsum  += zs;
            }
            val += sA[p][0] * Ysc[p] * Zsc[p] * inner;   // out1 path
            val += sA[p][2] * Zsc[p] * zsum;             // out3 path
        }
        WT[(size_t)(j * M + m) * KN + (k * N + n)] = f32_to_bf16(val);
    }
}

// ---------------------------------------------------------------------------
// Kernel 2: X f32 -> bf16 (so GEMM staging is a pure copy -> async-to-LDS)
// ---------------------------------------------------------------------------
__global__ __launch_bounds__(256)
void convert_x(const float* __restrict__ X, unsigned short* __restrict__ Xbf)
{
    const size_t t = (size_t)blockIdx.x * 256 + threadIdx.x;   // 8 floats each
    const float4* src = (const float4*)(X + t * 8);
    float4 a = src[0], b = src[1];
    uint4 pk;
    pk.x = (unsigned)f32_to_bf16(a.x) | ((unsigned)f32_to_bf16(a.y) << 16);
    pk.y = (unsigned)f32_to_bf16(a.z) | ((unsigned)f32_to_bf16(a.w) << 16);
    pk.z = (unsigned)f32_to_bf16(b.x) | ((unsigned)f32_to_bf16(b.y) << 16);
    pk.w = (unsigned)f32_to_bf16(b.z) | ((unsigned)f32_to_bf16(b.w) << 16);
    *(uint4*)(Xbf + t * 8) = pk;
}

// ---------------------------------------------------------------------------
// Kernel 3: OUT[512x4096] = Xbf * WT^T + bias; bf16 WMMA, f32 acc,
// double-buffered LDS fed by GLOBAL_LOAD_ASYNC_TO_LDS_B128.
// ---------------------------------------------------------------------------
constexpr int BT  = 128;   // block tile (rows == cols)
constexpr int KC  = 64;    // K chunk in LDS
constexpr int LDP = 72;    // padded LDS stride; 144B row pitch (16B aligned)
constexpr int BUFSZ = BT * LDP;

__global__ __launch_bounds__(128)
void gemm_bf16(const unsigned short* __restrict__ Xbf,
               const unsigned short* __restrict__ WT,
               const float* __restrict__ bias,
               float* __restrict__ OUT)
{
    __shared__ unsigned short Xs[2][BUFSZ];
    __shared__ unsigned short Ws[2][BUFSZ];

    const int tid  = threadIdx.x;
    const int lane = tid & 31;
    const int wave = tid >> 5;
    const int wr = wave >> 1, wc = wave & 1;
    const int ln16 = lane & 15;
    const int hi   = lane >> 4;

    const int rowBase = blockIdx.y * BT;
    const int colBase = blockIdx.x * BT;

    v8f acc[4][4];
    #pragma unroll
    for (int i = 0; i < 4; ++i)
        #pragma unroll
        for (int c = 0; c < 4; ++c) acc[i][c] = (v8f){0,0,0,0,0,0,0,0};

    const unsigned short* xg = Xbf + (size_t)(rowBase + tid) * KN;
    const unsigned short* wg = WT  + (size_t)(colBase + tid) * KN;
    unsigned short* xl = &Xs[0][tid * LDP];
    unsigned short* wl = &Ws[0][tid * LDP];

    constexpr int NCHUNK = KN / KC;   // 64

    // prologue: fill buffer 0 (16 async b128 per thread)
    stage_chunk(xg, xl);
    stage_chunk(wg, wl);

    for (int i = 0; i < NCHUNK; ++i) {
        const int cur = i & 1;
        if (i + 1 < NCHUNK) {
            const int nxt = cur ^ 1;
            stage_chunk(xg + (i + 1) * KC, xl + nxt * BUFSZ);
            stage_chunk(wg + (i + 1) * KC, wl + nxt * BUFSZ);
            WAIT_ASYNC(16);     // in-order completion => buffer `cur` resident
        } else {
            WAIT_ASYNC(0);
        }
        __syncthreads();

        const unsigned short* Xb = &Xs[cur][0];
        const unsigned short* Wb = &Ws[cur][0];
        #pragma unroll
        for (int ks = 0; ks < KC; ks += 32) {
            V16 a[4], b[4];
            const int akb = ks + hi * 16;   // A 16x32: lane<16 -> K 0..15
            #pragma unroll
            for (int rt = 0; rt < 4; ++rt) {
                const unsigned short* pr = Xb + (wr * 64 + rt * 16 + ln16) * LDP;
                a[rt].h[0] = *(const v8bf*)(pr + akb);
                a[rt].h[1] = *(const v8bf*)(pr + akb + 8);
            }
            const int bkb = ks + hi * 16;   // B 32x16: lane<16 -> K 0..15
            #pragma unroll
            for (int ct = 0; ct < 4; ++ct) {
                const unsigned short* pc = Wb + (wc * 64 + ct * 16 + ln16) * LDP;
                b[ct].h[0] = *(const v8bf*)(pc + bkb);
                b[ct].h[1] = *(const v8bf*)(pc + bkb + 8);
            }
            #pragma unroll
            for (int rt = 0; rt < 4; ++rt)
                #pragma unroll
                for (int ct = 0; ct < 4; ++ct)
                    acc[rt][ct] = __builtin_amdgcn_wmma_f32_16x16x32_bf16(
                        false, a[rt].v, false, b[ct].v,
                        (short)0, acc[rt][ct], false, false);
        }
        __syncthreads();   // safe to overwrite buffer `cur` next iteration
    }

    // epilogue: C/D layout row = base + g + hi*8, col = base + ln16.
    // Output written once, never re-read -> nontemporal stores.
    #pragma unroll
    for (int ct = 0; ct < 4; ++ct) {
        const int col = colBase + wc * 64 + ct * 16 + ln16;
        const float bv = bias[col];
        #pragma unroll
        for (int rt = 0; rt < 4; ++rt) {
            const int r0 = rowBase + wr * 64 + rt * 16 + hi * 8;
            #pragma unroll
            for (int g = 0; g < 8; ++g)
                __builtin_nontemporal_store(acc[rt][ct][g] + bv,
                                            &OUT[(size_t)(r0 + g) * JM + col]);
        }
    }
}

// ---------------------------------------------------------------------------
extern "C" void kernel_launch(void* const* d_in, const int* in_sizes, int n_in,
                              void* d_out, int out_size, void* d_ws, size_t ws_size,
                              hipStream_t stream) {
    const float* x    = (const float*)d_in[0];
    const float* Y_fp = (const float*)d_in[1];
    const float* Z_fp = (const float*)d_in[2];
    const float* A    = (const float*)d_in[3];
    const float* bias = (const float*)d_in[4];
    float* out = (float*)d_out;

    // scratch: WT = 4096*4096 bf16 (32 MB), Xbf = 512*4096 bf16 (4 MB)
    unsigned short* WT  = (unsigned short*)d_ws;
    unsigned short* Xbf = WT + (size_t)JM * KN;

    build_w<<<dim3(K, J), 256, 0, stream>>>(Y_fp, Z_fp, A, WT);
    convert_x<<<dim3((BATCH * KN) / 8 / 256), 256, 0, stream>>>(x, Xbf);
    gemm_bf16<<<dim3(JM / BT, BATCH / BT), 128, 0, stream>>>(Xbf, WT, bias, out);
}
